// BiMatchLSTM_64072322121775
// MI455X (gfx1250) — matrix operand order repaired and checked by reading, not compiled, not run on hardware
//
#include <hip/hip_runtime.h>
#include <hip/hip_bf16.h>
#include <math.h>

// ---------------------------------------------------------------------------
// BiMatchLSTM for MI455X (gfx1250, wave32, WMMA bf16)
// B=32, TP=256, TQ=64, D=256, H=128
// ---------------------------------------------------------------------------

#define B_  32
#define TP_ 256
#define TQ_ 64
#define D_  256
#define H_  128
#define NL_ 512   // 4*H lstm gate width

typedef __bf16 bf16_t;
typedef __attribute__((ext_vector_type(16))) __bf16 v16bf;
typedef __attribute__((ext_vector_type(8)))  float  v8f;

__device__ __forceinline__ bf16_t f2bf(float f) {
  unsigned u = __float_as_uint(f);
  u += 0x7FFFu + ((u >> 16) & 1u);              // round to nearest even
  unsigned short s = (unsigned short)(u >> 16);
  return __builtin_bit_cast(bf16_t, s);
}
__device__ __forceinline__ float bf2f(bf16_t h) {
  unsigned short s = __builtin_bit_cast(unsigned short, h);
  return __uint_as_float(((unsigned)s) << 16);
}
__device__ __forceinline__ float sigmoidf_(float x) {
  return 1.0f / (1.0f + __expf(-x));
}

// A fragment (16xK tile, row-major, stride ldk). ISA 16-bit A layout:
// lane L: M=L%16; lane-half hh=L/16 holds K = hh*8..hh*8+7 (VGPR0-3) and
// K = hh*8+16..hh*8+23 (VGPR4-7)  => two contiguous 8-element chunks.
__device__ __forceinline__ v16bf load_frag_a(const bf16_t* tile, int ldk,
                                             int k0, int lane) {
  int row = lane & 15, hh = lane >> 4;
  const bf16_t* p = tile + (size_t)row * ldk + k0 + hh * 8;
  v16bf r;
#pragma unroll
  for (int i = 0; i < 8; ++i) { r[i] = p[i]; r[i + 8] = p[i + 16]; }
  return r;
}

// B fragment from N-major (transposed) weights Bt[N][K]. ISA 16-bit B layout:
// lane L: N=L%16; lane-half hh=L/16 holds K = hh*8..hh*8+7 and hh*8+16..hh*8+23.
__device__ __forceinline__ v16bf load_frag_b(const bf16_t* tileT, int ldk,
                                             int k0, int lane) {
  int col = lane & 15, hh = lane >> 4;
  const bf16_t* p = tileT + (size_t)col * ldk + k0 + hh * 8;
  v16bf r;
#pragma unroll
  for (int i = 0; i < 8; ++i) { r[i] = p[i]; r[i + 8] = p[i + 16]; }
  return r;
}

__device__ __forceinline__ v8f wmma_bf16(v16bf a, v16bf b, v8f c) {
  return __builtin_amdgcn_wmma_f32_16x16x32_bf16(false, a, false, b,
                                                 (short)0, c, false, false);
}

// ---------------------------------------------------------------------------
// Elementwise f32 -> bf16
// ---------------------------------------------------------------------------
__global__ void conv_bf16_kernel(const float* __restrict__ in,
                                 bf16_t* __restrict__ out, int n) {
  int i = blockIdx.x * blockDim.x + threadIdx.x;
  if (i < n) out[i] = f2bf(in[i]);
}

// Transpose + convert: out[n*K+k] = in[(row0+k)*Nin + perm(n)], optional gate
// interleave perm(n) = (n&3)*H + (n>>2) for the 512-wide LSTM weights.
__global__ void transpose_conv_kernel(const float* __restrict__ in,
                                      bf16_t* __restrict__ out, int K, int N,
                                      int row0, int Nin, int permute) {
  int i = blockIdx.x * blockDim.x + threadIdx.x;
  if (i >= N * K) return;
  int n = i / K, k = i - n * K;
  int col = permute ? ((n & 3) * H_ + (n >> 2)) : n;
  out[(size_t)n * K + k] = f2bf(in[(size_t)(row0 + k) * Nin + col]);
}

__global__ void permute_bias_kernel(const float* __restrict__ in,
                                    float* __restrict__ out) {
  int n = blockIdx.x * blockDim.x + threadIdx.x;
  if (n < NL_) out[n] = in[(n & 3) * H_ + (n >> 2)];
}

// ---------------------------------------------------------------------------
// Precompute GEMM: C[M,N] f32 = A_bf16[M,K] @ Bt_bf16[N,K]^T + bias[N]
// One 16x16 tile per wave, v_wmma_f32_16x16x32_bf16 over K.
// ---------------------------------------------------------------------------
__global__ __launch_bounds__(256)
void gemm_wmma_kernel(const bf16_t* __restrict__ A,
                      const bf16_t* __restrict__ Bt,
                      const float* __restrict__ bias, float* __restrict__ C,
                      int M, int N, int K) {
  int wave = (blockIdx.x * 256 + threadIdx.x) >> 5;
  int lane = threadIdx.x & 31;
  int ntn = N >> 4;
  int total = (M >> 4) * ntn;
  if (wave >= total) return;               // uniform per wave: EXEC stays full
  int mt = wave / ntn, nt = wave - mt * ntn;
  const bf16_t* aT = A + (size_t)mt * 16 * K;
  const bf16_t* bT = Bt + (size_t)nt * 16 * K;
  v8f acc = {0.f, 0.f, 0.f, 0.f, 0.f, 0.f, 0.f, 0.f};
  for (int k = 0; k < K; k += 32) {
    v16bf af = load_frag_a(aT, K, k, lane);
    v16bf bf_ = load_frag_b(bT, K, k, lane);
    acc = wmma_bf16(af, bf_, acc);
  }
  int nl = lane & 15, hh = lane >> 4;
  int n = nt * 16 + nl;
  float bv = bias[n];
  int mb = mt * 16 + hh * 8;
#pragma unroll
  for (int i = 0; i < 8; ++i)
    C[(size_t)(mb + i) * N + n] = acc[i] + bv;
}

// ---------------------------------------------------------------------------
// Persistent recurrent kernel: one block per direction, 1024 threads = 32 waves.
// LDS: h (bf16), c (f32), gpr (f32), att (f32), z (bf16)  == 64 KB.
// ---------------------------------------------------------------------------
__global__ __launch_bounds__(1024, 1)
void rnn_kernel(const float* __restrict__ Xpr,      // [B*TP][H]   x@Wpr_x+b_pr
                const float* __restrict__ Xlf,      // [B*TP][NL]  permuted
                const float* __restrict__ Xlb,
                const bf16_t* __restrict__ Wt_prh,  // [H][H]
                const bf16_t* __restrict__ Wt_lfz,  // [NL][D] permuted
                const bf16_t* __restrict__ Wt_lbz,
                const bf16_t* __restrict__ Wt_lfh,  // [NL][H] permuted
                const bf16_t* __restrict__ Wt_lbh,
                const float* __restrict__ Gq,       // [B*TQ][H]
                const float* __restrict__ input_q,  // [B*TQ][D]
                const float* __restrict__ mask_p,   // [B][TP]
                const float* __restrict__ mask_q,   // [B][TQ]
                const float* __restrict__ w_att,    // [H]
                const float* __restrict__ b_match,  // scalar
                float* __restrict__ out_states,     // [B][TP][2H]
                float* __restrict__ out_last) {     // [B][2H]
  __shared__ float  c_sh[B_][H_];     // 16 KB
  __shared__ bf16_t h_sh[B_][H_];     //  8 KB
  __shared__ float  gpr_sh[B_][H_];   // 16 KB
  __shared__ float  att_sh[B_][TQ_];  //  8 KB
  __shared__ bf16_t z_sh[B_][D_];     // 16 KB

  const int dir = blockIdx.x;
  const float* Xl = dir ? Xlb : Xlf;
  const bf16_t* Wlz = dir ? Wt_lbz : Wt_lfz;
  const bf16_t* Wlh = dir ? Wt_lbh : Wt_lfh;

  const int tid = threadIdx.x;
  const int wv = tid >> 5;
  const int lane = tid & 31;
  const float bm = b_match[0];

  for (int i = tid; i < B_ * H_; i += 1024) {
    int b = i >> 7, j = i & (H_ - 1);
    c_sh[b][j] = 0.f;
    h_sh[b][j] = f2bf(0.f);
  }
  __syncthreads();

  for (int s = 0; s < TP_; ++s) {
    const int t = dir ? (TP_ - 1 - s) : s;

    // ---- phase 1: gpr = Xpr[t] + h @ Wpr_h  (M=32,N=128,K=128), 16 tiles ---
    if (wv < 16) {
      int mt = wv >> 3, nt = wv & 7;
      int nl = lane & 15, hh = lane >> 4;
      int n = nt * 16 + nl;
      int mb = mt * 16 + hh * 8;
      v8f acc;
#pragma unroll
      for (int i = 0; i < 8; ++i)
        acc[i] = Xpr[(size_t)((mb + i) * TP_ + t) * H_ + n];
#pragma unroll
      for (int k = 0; k < H_; k += 32) {
        v16bf af = load_frag_a(&h_sh[mt * 16][0], H_, k, lane);
        v16bf bf_ = load_frag_b(Wt_prh + (size_t)nt * 16 * H_, H_, k, lane);
        acc = wmma_bf16(af, bf_, acc);
      }
#pragma unroll
      for (int i = 0; i < 8; ++i) gpr_sh[mb + i][n] = acc[i];
    }
    __syncthreads();

    // ---- phase 2: a[b,tq] = sum_h tanh(gpr+Gq)*w_att + b_match ------------
    for (int p = tid; p < B_ * TQ_; p += 1024) {
      int b = p >> 6;
      const float* gq = Gq + (size_t)p * H_;
      float sum = 0.f;
      for (int hI = 0; hI < H_; ++hI)
        sum += tanhf(gpr_sh[b][hI] + gq[hI]) * w_att[hI];
      float a = sum + bm;
      a = fminf(15.f, fmaxf(-15.f, a)) * mask_q[p];
      att_sh[b][p & (TQ_ - 1)] = a;
    }
    __syncthreads();

    // ---- phase 3: masked softmax over TQ, one wave per batch --------------
    {
      int b = wv;
      float a0 = att_sh[b][lane], a1 = att_sh[b][lane + 32];
      float mx = fmaxf(a0, a1);
#pragma unroll
      for (int o = 16; o; o >>= 1) mx = fmaxf(mx, __shfl_xor(mx, o, 32));
      float m0 = mask_q[b * TQ_ + lane], m1 = mask_q[b * TQ_ + lane + 32];
      float e0 = __expf(a0 - mx) * m0, e1 = __expf(a1 - mx) * m1;
      float sm = e0 + e1;
#pragma unroll
      for (int o = 16; o; o >>= 1) sm += __shfl_xor(sm, o, 32);
      float inv = 1.f / (sm + 1e-6f);
      att_sh[b][lane] = e0 * inv;
      att_sh[b][lane + 32] = e1 * inv;
    }
    __syncthreads();

    // ---- phase 4: z[b,d] = sum_tq alpha * input_q -------------------------
    {
      int b = tid >> 5;
      int d0 = (tid & 31) * 8;
      float acc[8];
#pragma unroll
      for (int i = 0; i < 8; ++i) acc[i] = 0.f;
      const float* qb = input_q + ((size_t)b * TQ_) * D_ + d0;
      for (int tq = 0; tq < TQ_; ++tq) {
        float al = att_sh[b][tq];
        const float* qr = qb + (size_t)tq * D_;
#pragma unroll
        for (int i = 0; i < 8; ++i) acc[i] += al * qr[i];
      }
#pragma unroll
      for (int i = 0; i < 8; ++i) z_sh[b][d0 + i] = f2bf(acc[i]);
    }
    __syncthreads();

    // ---- phase 5: pre = Xl[t] + z@Wz + h@Wh (gate-interleaved columns) ----
    // 64 tiles (2 M-tiles x 32 N-tiles), 2 tiles per wave, kept in registers.
    v8f pacc[2];
#pragma unroll
    for (int u = 0; u < 2; ++u) {
      int tile = wv * 2 + u;
      int mt = tile >> 5, nt = tile & 31;
      int nl = lane & 15, hh = lane >> 4;
      int n = nt * 16 + nl;
      int mb = mt * 16 + hh * 8;
      v8f acc;
#pragma unroll
      for (int i = 0; i < 8; ++i)
        acc[i] = Xl[(size_t)((mb + i) * TP_ + t) * NL_ + n];
#pragma unroll
      for (int k = 0; k < D_; k += 32) {
        v16bf af = load_frag_a(&z_sh[mt * 16][0], D_, k, lane);
        v16bf bf_ = load_frag_b(Wlz + (size_t)nt * 16 * D_, D_, k, lane);
        acc = wmma_bf16(af, bf_, acc);
      }
#pragma unroll
      for (int k = 0; k < H_; k += 32) {
        v16bf af = load_frag_a(&h_sh[mt * 16][0], H_, k, lane);
        v16bf bf_ = load_frag_b(Wlh + (size_t)nt * 16 * H_, H_, k, lane);
        acc = wmma_bf16(af, bf_, acc);
      }
      pacc[u] = acc;
    }
    __syncthreads();  // all waves done reading h_sh / z_sh

    // ---- phase 6: gate combine (f,i,o,g live in 4 adjacent lanes) ---------
#pragma unroll
    for (int u = 0; u < 2; ++u) {
      int tile = wv * 2 + u;
      int mt = tile >> 5, nt = tile & 31;
      int nl = lane & 15, hh = lane >> 4;
      int g = nl & 3;
      int j = nt * 4 + (nl >> 2);
      int mb = mt * 16 + hh * 8;
      int base = lane & ~3;
#pragma unroll
      for (int i = 0; i < 8; ++i) {
        float fv = __shfl(pacc[u][i], base + 0, 32);
        float iv = __shfl(pacc[u][i], base + 1, 32);
        float ov = __shfl(pacc[u][i], base + 2, 32);
        float gv = __shfl(pacc[u][i], base + 3, 32);
        int b = mb + i;
        float m = mask_p[(size_t)b * TP_ + t];
        float cold = c_sh[b][j];
        float hold = bf2f(h_sh[b][j]);
        float c1 = sigmoidf_(fv) * cold + sigmoidf_(iv) * tanhf(gv);
        c1 = c1 * m + cold * (1.f - m);
        float h1 = sigmoidf_(ov) * tanhf(c1);
        h1 = h1 * m + hold * (1.f - m);
        if (g == 0) {
          c_sh[b][j] = c1;
          h_sh[b][j] = f2bf(h1);
          out_states[((size_t)b * TP_ + t) * (2 * H_) + dir * H_ + j] = h1 * m;
        }
      }
    }
    __syncthreads();
  }

  for (int i = tid; i < B_ * H_; i += 1024) {
    int b = i >> 7, j = i & (H_ - 1);
    out_last[(size_t)b * (2 * H_) + dir * H_ + j] = bf2f(h_sh[b][j]);
  }
}

// ---------------------------------------------------------------------------
extern "C" void kernel_launch(void* const* d_in, const int* in_sizes, int n_in,
                              void* d_out, int out_size, void* d_ws,
                              size_t ws_size, hipStream_t stream) {
  (void)in_sizes; (void)n_in; (void)out_size; (void)ws_size;
  const float* input_p = (const float*)d_in[0];
  const float* mask_p  = (const float*)d_in[1];
  const float* input_q = (const float*)d_in[2];
  const float* mask_q  = (const float*)d_in[3];
  const float* W_pr    = (const float*)d_in[4];
  const float* b_pr    = (const float*)d_in[5];
  const float* W_q     = (const float*)d_in[6];
  const float* b_q     = (const float*)d_in[7];
  const float* w_att   = (const float*)d_in[8];
  const float* b_match = (const float*)d_in[9];
  const float* W_lf    = (const float*)d_in[10];
  const float* b_lf    = (const float*)d_in[11];
  const float* W_lb    = (const float*)d_in[12];
  const float* b_lb    = (const float*)d_in[13];

  float* out_states = (float*)d_out;
  float* out_last = out_states + (size_t)B_ * TP_ * 2 * H_;

  char* ws = (char*)d_ws;
  size_t off = 0;
  auto alloc = [&](size_t bytes) -> char* {
    char* p = ws + off;
    off += (bytes + 255) & ~(size_t)255;
    return p;
  };
  bf16_t* bf_p   = (bf16_t*)alloc((size_t)B_ * TP_ * D_ * 2);
  bf16_t* bf_q   = (bf16_t*)alloc((size_t)B_ * TQ_ * D_ * 2);
  bf16_t* Wt_prx = (bf16_t*)alloc((size_t)H_ * D_ * 2);
  bf16_t* Wt_prh = (bf16_t*)alloc((size_t)H_ * H_ * 2);
  bf16_t* Wt_q   = (bf16_t*)alloc((size_t)H_ * D_ * 2);
  bf16_t* Wt_lfx = (bf16_t*)alloc((size_t)NL_ * D_ * 2);
  bf16_t* Wt_lfz = (bf16_t*)alloc((size_t)NL_ * D_ * 2);
  bf16_t* Wt_lfh = (bf16_t*)alloc((size_t)NL_ * H_ * 2);
  bf16_t* Wt_lbx = (bf16_t*)alloc((size_t)NL_ * D_ * 2);
  bf16_t* Wt_lbz = (bf16_t*)alloc((size_t)NL_ * D_ * 2);
  bf16_t* Wt_lbh = (bf16_t*)alloc((size_t)NL_ * H_ * 2);
  float* blf = (float*)alloc(NL_ * 4);
  float* blb = (float*)alloc(NL_ * 4);
  float* Gq  = (float*)alloc((size_t)B_ * TQ_ * H_ * 4);
  float* Xpr = (float*)alloc((size_t)B_ * TP_ * H_ * 4);
  float* Xlf = (float*)alloc((size_t)B_ * TP_ * NL_ * 4);
  float* Xlb = (float*)alloc((size_t)B_ * TP_ * NL_ * 4);

  // --- converts ---
  {
    int n = B_ * TP_ * D_;
    conv_bf16_kernel<<<(n + 255) / 256, 256, 0, stream>>>(input_p, bf_p, n);
  }
  {
    int n = B_ * TQ_ * D_;
    conv_bf16_kernel<<<(n + 255) / 256, 256, 0, stream>>>(input_q, bf_q, n);
  }
  auto tlaunch = [&](const float* in, bf16_t* out, int K, int N, int row0,
                     int Nin, int perm) {
    int n = N * K;
    transpose_conv_kernel<<<(n + 255) / 256, 256, 0, stream>>>(in, out, K, N,
                                                               row0, Nin, perm);
  };
  tlaunch(W_pr, Wt_prx, D_, H_, 0, H_, 0);
  tlaunch(W_pr, Wt_prh, H_, H_, D_, H_, 0);
  tlaunch(W_q,  Wt_q,   D_, H_, 0, H_, 0);
  tlaunch(W_lf, Wt_lfx, D_, NL_, 0,      NL_, 1);
  tlaunch(W_lf, Wt_lfz, D_, NL_, D_,     NL_, 1);
  tlaunch(W_lf, Wt_lfh, H_, NL_, 2 * D_, NL_, 1);
  tlaunch(W_lb, Wt_lbx, D_, NL_, 0,      NL_, 1);
  tlaunch(W_lb, Wt_lbz, D_, NL_, D_,     NL_, 1);
  tlaunch(W_lb, Wt_lbh, H_, NL_, 2 * D_, NL_, 1);
  permute_bias_kernel<<<2, 256, 0, stream>>>(b_lf, blf);
  permute_bias_kernel<<<2, 256, 0, stream>>>(b_lb, blb);

  // --- precompute GEMMs (WMMA) ---
  auto gemm = [&](const bf16_t* A, const bf16_t* Bt, const float* bias,
                  float* C, int M, int N, int K) {
    int tiles = (M / 16) * (N / 16);
    gemm_wmma_kernel<<<(tiles + 7) / 8, 256, 0, stream>>>(A, Bt, bias, C, M, N, K);
  };
  gemm(bf_q, Wt_q,   b_q,  Gq,  B_ * TQ_, H_,  D_);
  gemm(bf_p, Wt_prx, b_pr, Xpr, B_ * TP_, H_,  D_);
  gemm(bf_p, Wt_lfx, blf,  Xlf, B_ * TP_, NL_, D_);
  gemm(bf_p, Wt_lbx, blb,  Xlb, B_ * TP_, NL_, D_);

  // --- recurrent scan: one block per direction ---
  rnn_kernel<<<2, 1024, 0, stream>>>(Xpr, Xlf, Xlb, Wt_prh, Wt_lfz, Wt_lbz,
                                     Wt_lfh, Wt_lbh, Gq, input_q, mask_p,
                                     mask_q, w_att, b_match, out_states,
                                     out_last);
}
